// DeBERTa_MoE_Model_91164975824953
// MI455X (gfx1250) — matrix operand order repaired and checked by reading, compiled, not run on hardware
//
#include <hip/hip_runtime.h>
#include <cstdint>
#include <cstddef>

// ---------------- problem constants ----------------
#define NTOK   4096          // B*S = 8*512
#define HD     768
#define FD     3072
#define NE     6
#define TK     2
#define NPAIR  (NTOK*TK)     // 8192

typedef __bf16 bf16;
typedef __attribute__((ext_vector_type(16))) __bf16 v16bf;
typedef __attribute__((ext_vector_type(8)))  __bf16 v8bf;
typedef __attribute__((ext_vector_type(8)))  float  v8f;

// fragment union: two contiguous 16-byte loads fill the 8-VGPR WMMA operand
union Frag {
    v16bf v;
    v8bf  h[2];
};

// ---------------- fp32 -> bf16 convert (no transpose, for x) ----------------
__global__ void cvt_f32_bf16(const float* __restrict__ s, bf16* __restrict__ d, size_t n) {
    size_t i = (size_t)blockIdx.x * blockDim.x + threadIdx.x;
    size_t stride = (size_t)gridDim.x * blockDim.x;
    for (; i < n; i += stride) d[i] = (bf16)s[i];
}

// ------- fp32 -> bf16 convert + transpose: src [E][R][C] -> dst [E][C][R] -------
__global__ __launch_bounds__(256) void cvt_transpose_bf16(
    const float* __restrict__ src, bf16* __restrict__ dst, int R, int C) {
    __shared__ __align__(16) bf16 tile[64][66];   // 66: odd 2-elem pad -> conflict-free
    const int e  = blockIdx.z;
    src += (size_t)e * R * C;
    dst += (size_t)e * R * C;
    const int c0 = blockIdx.x * 64;
    const int r0 = blockIdx.y * 64;
    const int tx = threadIdx.x & 63;
    const int ty = threadIdx.x >> 6;        // 0..3
#pragma unroll
    for (int rr = ty; rr < 64; rr += 4)
        tile[rr][tx] = (bf16)src[(size_t)(r0 + rr) * C + (c0 + tx)];
    __syncthreads();
#pragma unroll
    for (int cc = ty; cc < 64; cc += 4)
        dst[(size_t)(c0 + cc) * R + (r0 + tx)] = tile[tx][cc];
}

// ---------------- gating: scores, top-2, softmax, counts ----------------
__global__ void gate_topk(const float* __restrict__ x, const float* __restrict__ gw,
                          const float* __restrict__ gb, int* __restrict__ tidx,
                          float* __restrict__ twt, int* __restrict__ counts) {
    int n = blockIdx.x * blockDim.x + threadIdx.x;
    if (n >= NTOK) return;
    float s[NE];
#pragma unroll
    for (int e = 0; e < NE; ++e) s[e] = gb[e];
    const float* xr = x + (size_t)n * HD;
    for (int h = 0; h < HD; ++h) {
        float xv = xr[h];
#pragma unroll
        for (int e = 0; e < NE; ++e) s[e] += xv * gw[h * NE + e];
    }
    // top-2, ties -> lowest index (matches jax.lax.top_k)
    int i0 = 0;
    for (int e = 1; e < NE; ++e) if (s[e] > s[i0]) i0 = e;
    int i1 = (i0 == 0) ? 1 : 0;
    for (int e = 0; e < NE; ++e) if (e != i0 && s[e] > s[i1]) i1 = e;
    float z  = expf(s[i1] - s[i0]);      // s[i0] >= s[i1]
    float w0 = 1.0f / (1.0f + z);
    float w1 = z * w0;
    tidx[2 * n + 0] = i0; tidx[2 * n + 1] = i1;
    twt [2 * n + 0] = w0; twt [2 * n + 1] = w1;
    atomicAdd(&counts[i0], 1);
    atomicAdd(&counts[i1], 1);
}

// ---------------- per-expert segment bases ----------------
__global__ void expert_bases(const int* __restrict__ counts, int* __restrict__ bases,
                             int* __restrict__ cursor) {
    if (blockIdx.x == 0 && threadIdx.x == 0) {
        int acc = 0;
        for (int e = 0; e < NE; ++e) { bases[e] = acc; acc += counts[e]; cursor[e] = 0; }
    }
}

// ---------------- scatter (token,k) pairs into expert segments ----------------
__global__ void scatter_pairs(const int* __restrict__ tidx, const float* __restrict__ twt,
                              const int* __restrict__ bases, int* __restrict__ cursor,
                              int* __restrict__ ptok, float* __restrict__ pwt) {
    int n = blockIdx.x * blockDim.x + threadIdx.x;
    if (n >= NTOK) return;
#pragma unroll
    for (int k = 0; k < TK; ++k) {
        int e = tidx[2 * n + k];
        int slot = bases[e] + atomicAdd(&cursor[e], 1);
        ptok[slot] = n;
        pwt [slot] = twt[2 * n + k];
    }
}

// ---------------- up projection: hmid = gelu(x @ w1[e] + b1[e]) ----------------
// one wave per block; 64 gathered token rows x 64 F cols; B from transposed weights
__global__ __launch_bounds__(32) void moe_up_wmma(
    const bf16* __restrict__ xbf, const bf16* __restrict__ w1t,  // [E][FD][HD]
    const float* __restrict__ b1, const int* __restrict__ ptok,
    const int* __restrict__ counts, const int* __restrict__ bases,
    bf16* __restrict__ hmid)
{
    const int e     = blockIdx.z;
    const int count = counts[e];
    const int rb    = blockIdx.y * 64;
    if (rb >= count) return;
    const int f0   = blockIdx.x * 64;
    const int lane = threadIdx.x;
    const int m    = lane & 15;          // A row / B col / C col within 16-tile
    const int klo  = (lane >> 4) * 8;    // unused (A fragment also loads at kh)
    const int kh   = (lane >> 4) * 16;   // B-fragment K half     (contiguous k = 16h + s; A too)
    const int b0   = bases[e];

    const bf16* arow[4];
#pragma unroll
    for (int mi = 0; mi < 4; ++mi) {
        int r = rb + mi * 16 + m; if (r >= count) r = count - 1;  // clamp; stores masked
        arow[mi] = xbf + (size_t)ptok[b0 + r] * HD;
    }
    const bf16* wbase = w1t + (size_t)e * FD * HD;

    v8f zero = {0.f,0.f,0.f,0.f,0.f,0.f,0.f,0.f};
    v8f acc[4][4];
#pragma unroll
    for (int mi = 0; mi < 4; ++mi)
#pragma unroll
        for (int ti = 0; ti < 4; ++ti) acc[mi][ti] = zero;

    for (int k0 = 0; k0 < HD; k0 += 32) {
        Frag a[4];
#pragma unroll
        for (int mi = 0; mi < 4; ++mi) {
            a[mi].h[0] = *(const v8bf*)(arow[mi] + k0 + kh);
            a[mi].h[1] = *(const v8bf*)(arow[mi] + k0 + kh + 8);
        }
        Frag b[4];
#pragma unroll
        for (int ti = 0; ti < 4; ++ti) {
            const bf16* bp = wbase + (size_t)(f0 + ti * 16 + m) * HD + k0 + kh;
            b[ti].h[0] = *(const v8bf*)(bp);
            b[ti].h[1] = *(const v8bf*)(bp + 8);
        }
#pragma unroll
        for (int mi = 0; mi < 4; ++mi)
#pragma unroll
            for (int ti = 0; ti < 4; ++ti)
                acc[mi][ti] = __builtin_amdgcn_wmma_f32_16x16x32_bf16(
                    false, a[mi].v, false, b[ti].v, (short)0, acc[mi][ti], false, false);
    }

    // epilogue: bias + exact gelu -> bf16 hmid
    const int mrow = (lane >> 4) * 8;
#pragma unroll
    for (int ti = 0; ti < 4; ++ti) {
        const int f = f0 + ti * 16 + m;
        const float bias = b1[e * FD + f];
#pragma unroll
        for (int mi = 0; mi < 4; ++mi) {
#pragma unroll
            for (int i = 0; i < 8; ++i) {
                const int rr = rb + mi * 16 + mrow + i;
                if (rr < count) {
                    float v = acc[mi][ti][i] + bias;
                    float g = 0.5f * v * (1.0f + erff(v * 0.70710678118654752f));
                    hmid[(size_t)(b0 + rr) * FD + f] = (bf16)g;
                }
            }
        }
    }
}

// ---------------- down projection + weighted combine ----------------
__global__ __launch_bounds__(32) void moe_down_wmma(
    const bf16* __restrict__ hmid, const bf16* __restrict__ w2t,  // [E][HD][FD]
    const float* __restrict__ b2, const int* __restrict__ ptok,
    const float* __restrict__ pwt, const int* __restrict__ counts,
    const int* __restrict__ bases, float* __restrict__ out)
{
    const int e     = blockIdx.z;
    const int count = counts[e];
    const int rb    = blockIdx.y * 64;
    if (rb >= count) return;
    const int n0   = blockIdx.x * 64;
    const int lane = threadIdx.x;
    const int m    = lane & 15;
    const int klo  = (lane >> 4) * 8;
    const int kh   = (lane >> 4) * 16;
    const int b0   = bases[e];

    const bf16* arow[4];
#pragma unroll
    for (int mi = 0; mi < 4; ++mi) {
        int r = rb + mi * 16 + m; if (r >= count) r = count - 1;
        arow[mi] = hmid + (size_t)(b0 + r) * FD;
    }
    const bf16* wbase = w2t + (size_t)e * HD * FD;

    v8f zero = {0.f,0.f,0.f,0.f,0.f,0.f,0.f,0.f};
    v8f acc[4][4];
#pragma unroll
    for (int mi = 0; mi < 4; ++mi)
#pragma unroll
        for (int ti = 0; ti < 4; ++ti) acc[mi][ti] = zero;

    for (int k0 = 0; k0 < FD; k0 += 32) {
        Frag a[4];
#pragma unroll
        for (int mi = 0; mi < 4; ++mi) {
            a[mi].h[0] = *(const v8bf*)(arow[mi] + k0 + kh);
            a[mi].h[1] = *(const v8bf*)(arow[mi] + k0 + kh + 8);
        }
        Frag b[4];
#pragma unroll
        for (int ti = 0; ti < 4; ++ti) {
            const bf16* bp = wbase + (size_t)(n0 + ti * 16 + m) * FD + k0 + kh;
            b[ti].h[0] = *(const v8bf*)(bp);
            b[ti].h[1] = *(const v8bf*)(bp + 8);
        }
#pragma unroll
        for (int mi = 0; mi < 4; ++mi)
#pragma unroll
            for (int ti = 0; ti < 4; ++ti)
                acc[mi][ti] = __builtin_amdgcn_wmma_f32_16x16x32_bf16(
                    false, a[mi].v, false, b[ti].v, (short)0, acc[mi][ti], false, false);
    }

    const int mrow = (lane >> 4) * 8;
    int   toks[4][8]; float wts[4][8]; bool valid[4][8];
#pragma unroll
    for (int mi = 0; mi < 4; ++mi)
#pragma unroll
        for (int i = 0; i < 8; ++i) {
            int rr = rb + mi * 16 + mrow + i;
            valid[mi][i] = (rr < count);
            int slot = b0 + (valid[mi][i] ? rr : 0);
            toks[mi][i] = ptok[slot];
            wts [mi][i] = pwt [slot];
        }
    // exactly 2 commutative atomic adds per output element onto zeroed base
#pragma unroll
    for (int ti = 0; ti < 4; ++ti) {
        const int h = n0 + ti * 16 + m;
        const float bias = b2[e * HD + h];
#pragma unroll
        for (int mi = 0; mi < 4; ++mi)
#pragma unroll
            for (int i = 0; i < 8; ++i)
                if (valid[mi][i])
                    atomicAdd(&out[(size_t)toks[mi][i] * HD + h],
                              wts[mi][i] * (acc[mi][ti][i] + bias));
    }
}

// ---------------- host launch ----------------
extern "C" void kernel_launch(void* const* d_in, const int* in_sizes, int n_in,
                              void* d_out, int out_size, void* d_ws, size_t ws_size,
                              hipStream_t stream) {
    const float* x      = (const float*)d_in[0];
    const float* gate_w = (const float*)d_in[1];
    const float* gate_b = (const float*)d_in[2];
    const float* w1     = (const float*)d_in[3];
    const float* b1     = (const float*)d_in[4];
    const float* w2     = (const float*)d_in[5];
    const float* b2     = (const float*)d_in[6];
    float* out = (float*)d_out;

    // workspace layout
    size_t o = 0;
    const size_t O_XBF  = o; o += (size_t)NTOK * HD * sizeof(bf16);
    const size_t O_W1T  = o; o += (size_t)NE * HD * FD * sizeof(bf16);
    const size_t O_W2T  = o; o += (size_t)NE * FD * HD * sizeof(bf16);
    const size_t O_HMID = o; o += (size_t)NPAIR * FD * sizeof(bf16);
    const size_t O_PTOK = o; o += (size_t)NPAIR * sizeof(int);
    const size_t O_PWT  = o; o += (size_t)NPAIR * sizeof(float);
    const size_t O_TIDX = o; o += (size_t)NTOK * TK * sizeof(int);
    const size_t O_TWT  = o; o += (size_t)NTOK * TK * sizeof(float);
    const size_t O_CNT  = o; o += (size_t)NE * sizeof(int);
    const size_t O_BASE = o; o += (size_t)NE * sizeof(int);
    const size_t O_CUR  = o; o += (size_t)NE * sizeof(int);
    if (ws_size < o) return;

    char* ws = (char*)d_ws;
    bf16*  xbf  = (bf16*)(ws + O_XBF);
    bf16*  w1t  = (bf16*)(ws + O_W1T);   // [E][FD][HD]
    bf16*  w2t  = (bf16*)(ws + O_W2T);   // [E][HD][FD]
    bf16*  hmid = (bf16*)(ws + O_HMID);
    int*   ptok = (int*)(ws + O_PTOK);
    float* pwt  = (float*)(ws + O_PWT);
    int*   tidx = (int*)(ws + O_TIDX);
    float* twt  = (float*)(ws + O_TWT);
    int*   cnt  = (int*)(ws + O_CNT);
    int*   base = (int*)(ws + O_BASE);
    int*   cur  = (int*)(ws + O_CUR);

    hipMemsetAsync(d_out, 0, (size_t)out_size * sizeof(float), stream);
    hipMemsetAsync(cnt, 0, 3 * NE * sizeof(int), stream);  // cnt, base, cur contiguous

    cvt_f32_bf16<<<512, 256, 0, stream>>>(x, xbf, (size_t)NTOK * HD);
    // w1 [E][HD][FD] -> w1t [E][FD][HD]
    {
        dim3 g(FD / 64, HD / 64, NE);
        cvt_transpose_bf16<<<g, 256, 0, stream>>>(w1, w1t, HD, FD);
    }
    // w2 [E][FD][HD] -> w2t [E][HD][FD]
    {
        dim3 g(HD / 64, FD / 64, NE);
        cvt_transpose_bf16<<<g, 256, 0, stream>>>(w2, w2t, FD, HD);
    }

    gate_topk<<<NTOK / 256, 256, 0, stream>>>(x, gate_w, gate_b, tidx, twt, cnt);
    expert_bases<<<1, 1, 0, stream>>>(cnt, base, cur);
    scatter_pairs<<<NTOK / 256, 256, 0, stream>>>(tidx, twt, base, cur, ptok, pwt);

    dim3 upGrid(FD / 64, NTOK / 64, NE);    // (48, 64, 6)
    moe_up_wmma<<<upGrid, 32, 0, stream>>>(xbf, w1t, b1, ptok, cnt, base, hmid);

    dim3 dnGrid(HD / 64, NTOK / 64, NE);    // (12, 64, 6)
    moe_down_wmma<<<dnGrid, 32, 0, stream>>>(hmid, w2t, b2, ptok, pwt, cnt, base, out);
}
